// Attention_8126078124292
// MI455X (gfx1250) — matrix operand order repaired and checked by reading, not compiled, not run on hardware
//
#include <hip/hip_runtime.h>

// ---------------------------------------------------------------------------
// Types for CDNA5 WMMA (wave32): v_wmma_f32_16x16x32_bf16
// ---------------------------------------------------------------------------
typedef __attribute__((ext_vector_type(16))) __bf16 v16bf;
typedef __attribute__((ext_vector_type(8)))  float  v8f;

union FragBF16 {
    v16bf v;
    unsigned short u[16];
};

__device__ __forceinline__ unsigned short f32_to_bf16(float f) {
    unsigned int u = __float_as_uint(f);
    unsigned int r = u + 0x7FFFu + ((u >> 16) & 1u);   // round-to-nearest-even
    return (unsigned short)(r >> 16);
}

// Problem constants
#define BB 4
#define SS 2048
#define DD 1024

// ---------------------------------------------------------------------------
// fp32 -> bf16 conversion (flat)
// ---------------------------------------------------------------------------
__global__ void cvt_f32_bf16(const float* __restrict__ src,
                             unsigned short* __restrict__ dst, int n) {
    int i = blockIdx.x * blockDim.x + threadIdx.x;
    if (i < n) dst[i] = f32_to_bf16(src[i]);
}

// fp32 [K,N] -> bf16 transposed [N,K]  (weights: makes GEMM B-loads contiguous)
__global__ void cvt_f32_bf16_T(const float* __restrict__ src,
                               unsigned short* __restrict__ dst, int K, int N) {
    int i = blockIdx.x * blockDim.x + threadIdx.x;
    if (i < K * N) {
        int k = i / N, n = i % N;
        dst[(unsigned long long)n * K + k] = f32_to_bf16(src[i]);
    }
}

// ---------------------------------------------------------------------------
// Register-blocked bf16 WMMA GEMM: C[m,n] = sum_k A[m,k]*Bt[n,k] (+ bias[n])
//   A : row-major [M, lda] bf16
//   Bt: row-major [N, ldb] bf16 (B transposed -> K-contiguous per lane)
//   Block = 256 thr = 8 waves, block tile 128x256; wave tile 64x64
//   (4 M-frags x 4 N-frags = 16 WMMA per K-step; 16 b128 loads -> ratio 1.0).
//   grid.x = M/128, grid.y = N/256, grid.z = batch
//   causal!=0: per-wave K truncation at the wave's last row (attn @ V)
//   OUT_F32: fp32 output; TRANS_OUT: write C^T [N, ldc] (for V^T)
// ---------------------------------------------------------------------------
template <int OUT_F32, int TRANS_OUT>
__global__ __launch_bounds__(256)
void gemm_wmma_rb(const unsigned short* __restrict__ A, unsigned long long strideA, int lda,
                  const unsigned short* __restrict__ Bt, unsigned long long strideB, int ldb,
                  const float* __restrict__ bias,
                  void* __restrict__ Cout, unsigned long long strideC, int ldc,
                  int Ktiles, int causal) {
    const int w    = threadIdx.x >> 5;
    const int lane = threadIdx.x & 31;
    const int half = lane >> 4;
    const int lid  = lane & 15;
    const int wm   = w & 1;          // 2 M sub-blocks of 64
    const int wn   = w >> 1;         // 4 N sub-blocks of 64

    const int m0 = blockIdx.x * 128 + wm * 64;
    const int n0 = blockIdx.y * 256 + wn * 64;

    const unsigned short* Ab  = A  + strideA * (unsigned long long)blockIdx.z;
    const unsigned short* Btb = Bt + strideB * (unsigned long long)blockIdx.z;

    int ktEnd = Ktiles;
    if (causal) {
        int lim = (m0 + 64 + 31) / 32;          // keys needed: 0 .. m0+63
        ktEnd = (lim < Ktiles) ? lim : Ktiles;
    }

    v8f acc[4][4];
#pragma unroll
    for (int i = 0; i < 4; ++i)
#pragma unroll
        for (int j = 0; j < 4; ++j) acc[i][j] = (v8f){};

    for (int kt = 0; kt < ktEnd; ++kt) {
        const int k0 = kt * 32;
        FragBF16 a[4], b[4];
#pragma unroll
        for (int i = 0; i < 4; ++i) {
            const unsigned short* arow =
                Ab + (unsigned long long)(m0 + 16 * i + lid) * lda + (k0 + half * 16);
#pragma unroll
            for (int e = 0; e < 8; ++e) a[i].u[e] = arow[e];
#pragma unroll
            for (int e = 0; e < 8; ++e) a[i].u[8 + e] = arow[8 + e];
            __builtin_prefetch(arow + 32, 0, 1);        // next K-chunk
        }
#pragma unroll
        for (int j = 0; j < 4; ++j) {
            const unsigned short* brow =
                Btb + (unsigned long long)(n0 + 16 * j + lid) * ldb + (k0 + half * 16);
#pragma unroll
            for (int e = 0; e < 16; ++e) b[j].u[e] = brow[e];
            __builtin_prefetch(brow + 32, 0, 1);
        }
#pragma unroll
        for (int i = 0; i < 4; ++i)
#pragma unroll
            for (int j = 0; j < 4; ++j)
                acc[i][j] = __builtin_amdgcn_wmma_f32_16x16x32_bf16(
                    false, a[i].v, false, b[j].v, (short)0, acc[i][j], false, false);
    }

#pragma unroll
    for (int i = 0; i < 4; ++i) {
#pragma unroll
        for (int j = 0; j < 4; ++j) {
            const int n  = n0 + 16 * j + lid;
            const float bi = bias ? bias[n] : 0.0f;
            if (TRANS_OUT) {
                unsigned short* C = (unsigned short*)Cout + strideC * (unsigned long long)blockIdx.z;
                unsigned short* crow = C + (unsigned long long)n * ldc + (m0 + 16 * i + half * 8);
#pragma unroll
                for (int e = 0; e < 8; ++e) crow[e] = f32_to_bf16(acc[i][j][e] + bi);
            } else if (OUT_F32) {
                float* C = (float*)Cout + strideC * (unsigned long long)blockIdx.z;
#pragma unroll
                for (int e = 0; e < 8; ++e) {
                    int m = m0 + 16 * i + half * 8 + e;
                    C[(unsigned long long)m * ldc + n] = acc[i][j][e] + bi;
                }
            } else {
                unsigned short* C = (unsigned short*)Cout + strideC * (unsigned long long)blockIdx.z;
#pragma unroll
                for (int e = 0; e < 8; ++e) {
                    int m = m0 + 16 * i + half * 8 + e;
                    C[(unsigned long long)m * ldc + n] = f32_to_bf16(acc[i][j][e] + bi);
                }
            }
        }
    }
}

// ---------------------------------------------------------------------------
// Fused scores + causal mask + softmax.
//   One workgroup per (batch, 16-query strip); 8 waves x 16 key-tiles each,
//   processed in groups of 4 sharing one Q fragment (4 WMMA / K-step).
//   16x2048 fp32 strip in dynamic LDS (128 KB of the WGP's 320 KB).
//   Output: attn probabilities, bf16, row-major [B, S, S].
// ---------------------------------------------------------------------------
__global__ __launch_bounds__(256)
void scores_softmax(const unsigned short* __restrict__ Qb,
                    const unsigned short* __restrict__ Kb,
                    unsigned short* __restrict__ attnb) {
    extern __shared__ float smem[];
    float* strip   = smem;                 // [16][2048]
    float* red     = smem + 16 * SS;       // [16][16] partials
    float* rowstat = red + 256;            // [0..15] max, [16..31] sum

    const int b  = blockIdx.y;
    const int q0 = blockIdx.x * 16;
    const int w    = threadIdx.x >> 5;
    const int lane = threadIdx.x & 31;
    const int half = lane >> 4;
    const int lid  = lane & 15;
    const float inv_std = 0.03125f;        // 1/sqrt(1024)
    const float NEG_INF = -__builtin_inff();

    const unsigned short* Qrow0 =
        Qb + (unsigned long long)(b * SS + q0) * DD;

    for (int g = 0; g < 4; ++g) {                       // 4 groups of 4 key tiles
        const int key0g = (w * 16 + g * 4) * 16;        // group covers 64 keys
        if (key0g > q0 + 15) {                          // fully beyond causal bound
#pragma unroll
            for (int t = 0; t < 4; ++t)
#pragma unroll
                for (int e = 0; e < 8; ++e)
                    strip[(half * 8 + e) * SS + key0g + 16 * t + lid] = NEG_INF;
            continue;
        }
        v8f acc[4];
#pragma unroll
        for (int t = 0; t < 4; ++t) acc[t] = (v8f){};

        for (int kt = 0; kt < DD / 32; ++kt) {
            const int k0 = kt * 32;
            FragBF16 a, kb[4];
            const unsigned short* arow = Qrow0 + (unsigned long long)lid * DD + (k0 + half * 16);
#pragma unroll
            for (int e = 0; e < 8; ++e) a.u[e] = arow[e];
#pragma unroll
            for (int e = 0; e < 8; ++e) a.u[8 + e] = arow[8 + e];
#pragma unroll
            for (int t = 0; t < 4; ++t) {
                // B = K^T: element (d, key) = K[key, d] -> contiguous per lane
                const unsigned short* krow =
                    Kb + (unsigned long long)(b * SS + key0g + 16 * t + lid) * DD + (k0 + half * 16);
#pragma unroll
                for (int e = 0; e < 16; ++e) kb[t].u[e] = krow[e];
            }
#pragma unroll
            for (int t = 0; t < 4; ++t)
                acc[t] = __builtin_amdgcn_wmma_f32_16x16x32_bf16(
                    false, a.v, false, kb[t].v, (short)0, acc[t], false, false);
        }
#pragma unroll
        for (int t = 0; t < 4; ++t)
#pragma unroll
            for (int e = 0; e < 8; ++e) {
                const int m   = half * 8 + e;           // local query row
                const int key = key0g + 16 * t + lid;
                float v = acc[t][e] * inv_std;
                if (key > q0 + m) v = NEG_INF;          // causal mask
                strip[m * SS + key] = v;
            }
    }
    __syncthreads();

    // Row softmax: 16 threads per row, 128 cols per thread.
    const int row   = threadIdx.x & 15;
    const int chunk = threadIdx.x >> 4;
    const int c0 = chunk * 128, c1 = c0 + 128;

    float mx = NEG_INF;
    for (int c = c0; c < c1; ++c) mx = fmaxf(mx, strip[row * SS + c]);
    red[row * 16 + chunk] = mx;
    __syncthreads();
    if (chunk == 0) {
        float m2 = NEG_INF;
        for (int j = 0; j < 16; ++j) m2 = fmaxf(m2, red[row * 16 + j]);
        rowstat[row] = m2;
    }
    __syncthreads();
    const float rm = rowstat[row];
    float s = 0.0f;
    for (int c = c0; c < c1; ++c) {
        float p = __expf(strip[row * SS + c] - rm);     // exp(-inf)=0
        strip[row * SS + c] = p;
        s += p;
    }
    red[row * 16 + chunk] = s;
    __syncthreads();
    if (chunk == 0) {
        float t = 0.0f;
        for (int j = 0; j < 16; ++j) t += red[row * 16 + j];
        rowstat[16 + row] = t;
    }
    __syncthreads();
    const float inv = 1.0f / rowstat[16 + row];
    unsigned short* arow =
        attnb + ((unsigned long long)b * SS + q0 + row) * SS;
    for (int c = c0; c < c1; ++c)
        arow[c] = f32_to_bf16(strip[row * SS + c] * inv);
}

// ---------------------------------------------------------------------------
// Host-side orchestration
// ---------------------------------------------------------------------------
extern "C" void kernel_launch(void* const* d_in, const int* in_sizes, int n_in,
                              void* d_out, int out_size, void* d_ws, size_t ws_size,
                              hipStream_t stream) {
    const float* x  = (const float*)d_in[0];
    // d_in[1] = mask (causal handled analytically; unused)
    const float* Wq = (const float*)d_in[2];
    const float* bq = (const float*)d_in[3];
    const float* Wk = (const float*)d_in[4];
    const float* bk = (const float*)d_in[5];
    const float* Wv = (const float*)d_in[6];
    const float* bv = (const float*)d_in[7];
    const float* Wo = (const float*)d_in[8];
    const float* bo = (const float*)d_in[9];
    float* out = (float*)d_out;

    const int nX = BB * SS * DD;   // 8,388,608
    const int nW = DD * DD;        // 1,048,576

    // Workspace layout
    char* ws = (char*)d_ws;
    unsigned short* xb  = (unsigned short*)(ws);                       // 16 MB [B*S, D]
    unsigned short* WqT = (unsigned short*)(ws + 16ull * 1024 * 1024); //  2 MB [N, K]
    unsigned short* WkT = (unsigned short*)(ws + 18ull * 1024 * 1024);
    unsigned short* WvT = (unsigned short*)(ws + 20ull * 1024 * 1024);
    unsigned short* WoT = (unsigned short*)(ws + 22ull * 1024 * 1024);
    unsigned short* Qb  = (unsigned short*)(ws + 24ull * 1024 * 1024); // 16 MB [B*S, D]
    unsigned short* Kb  = (unsigned short*)(ws + 40ull * 1024 * 1024); // 16 MB [B*S, D]
    unsigned short* Vt  = (unsigned short*)(ws + 56ull * 1024 * 1024); // 16 MB [D, B*S] (V^T)
    unsigned short* Ab  = (unsigned short*)(ws + 72ull * 1024 * 1024); // 32 MB attn [B,S,S]
    unsigned short* Cb  = (unsigned short*)(ws + 104ull * 1024 * 1024);// 16 MB ctx [B*S, D]

    // 1) conversions (weights transposed so GEMM B-loads are K-contiguous)
    cvt_f32_bf16  <<<dim3((nX + 255) / 256), 256, 0, stream>>>(x,  xb,  nX);
    cvt_f32_bf16_T<<<dim3((nW + 255) / 256), 256, 0, stream>>>(Wq, WqT, DD, DD);
    cvt_f32_bf16_T<<<dim3((nW + 255) / 256), 256, 0, stream>>>(Wk, WkT, DD, DD);
    cvt_f32_bf16_T<<<dim3((nW + 255) / 256), 256, 0, stream>>>(Wv, WvT, DD, DD);
    cvt_f32_bf16_T<<<dim3((nW + 255) / 256), 256, 0, stream>>>(Wo, WoT, DD, DD);

    // 2) Q/K/V projections: [8192, 1024] x [1024, 1024]
    {
        dim3 grid(BB * SS / 128, DD / 256, 1);   // (64, 4, 1)
        gemm_wmma_rb<0, 0><<<grid, 256, 0, stream>>>(xb, 0ull, DD, WqT, 0ull, DD, bq,
                                                     (void*)Qb, 0ull, DD, DD / 32, 0);
        gemm_wmma_rb<0, 0><<<grid, 256, 0, stream>>>(xb, 0ull, DD, WkT, 0ull, DD, bk,
                                                     (void*)Kb, 0ull, DD, DD / 32, 0);
        // V written transposed: Vt[d, b*S+s], ldc = B*S
        gemm_wmma_rb<0, 1><<<grid, 256, 0, stream>>>(xb, 0ull, DD, WvT, 0ull, DD, bv,
                                                     (void*)Vt, 0ull, BB * SS, DD / 32, 0);
    }

    // 3) Fused QK^T * inv_std + causal mask + softmax -> bf16 probs
    {
        dim3 grid(SS / 16, BB, 1);               // (128, 4)
        size_t smem = (size_t)(16 * SS + 256 + 32) * sizeof(float);  // ~129 KB
        scores_softmax<<<grid, 256, smem, stream>>>(Qb, Kb, Ab);
    }

    // 4) ctx = attn @ V  (B operand = V^T, K-contiguous; per-wave causal K cut)
    {
        dim3 grid(SS / 128, DD / 256, BB);       // (16, 4, 4)
        gemm_wmma_rb<0, 0><<<grid, 256, 0, stream>>>(
            Ab, (unsigned long long)SS * SS, SS,     // A: attn, lda=S
            Vt, (unsigned long long)SS,      BB * SS,// Bt: V^T + b*S, ldb=B*S
            nullptr,
            (void*)Cb, (unsigned long long)SS * DD, DD,
            SS / 32, /*causal=*/1);
    }

    // 5) out = ctx @ Wo + bo  (fp32 output)
    {
        dim3 grid(BB * SS / 128, DD / 256, 1);   // (64, 4, 1)
        gemm_wmma_rb<1, 0><<<grid, 256, 0, stream>>>(Cb, 0ull, DD, WoT, 0ull, DD, bo,
                                                     (void*)out, 0ull, DD, DD / 32, 0);
    }
}
